// AttentionBlock_79663053406305
// MI455X (gfx1250) — matrix operand order repaired and checked by reading, not compiled, not run on hardware
//
#include <hip/hip_runtime.h>
#include <hip/hip_bf16.h>

typedef __attribute__((ext_vector_type(16))) _Float16 v16h;
typedef __attribute__((ext_vector_type(2)))  _Float16 h2;
typedef __attribute__((ext_vector_type(8)))  float    v8f;
typedef int v4i __attribute__((vector_size(16)));

#define NB   16
#define NC   512
#define NH   4
#define HD   128
#define NT   1024
#define GSZ  16   // channels per group (512/32)

// ---------------------------------------------------------------------------
// Async global->LDS staging (gfx1250 ASYNCcnt path) with safe fallback.
// Probe result: builtin exists; param0 = v4i __device__* (global), param1 = LDS.
// ---------------------------------------------------------------------------
#if defined(__has_builtin)
#if __has_builtin(__builtin_amdgcn_global_load_async_to_lds_b128) && \
    __has_builtin(__builtin_amdgcn_s_wait_asynccnt)
#define HAVE_ASYNC_LDS 1
#endif
#endif

__device__ __forceinline__ void async_copy_b128(const _Float16* g, _Float16* l) {
#ifdef HAVE_ASYNC_LDS
  __builtin_amdgcn_global_load_async_to_lds_b128(
      (__attribute__((address_space(1))) v4i*)(_Float16*)g,
      (__attribute__((address_space(3))) v4i*)l, 0, 0);
#else
  *(uint4*)l = *(const uint4*)g;   // sync fallback: vmem load + ds_store_b128
#endif
}

__device__ __forceinline__ void wait_async() {
#ifdef HAVE_ASYNC_LDS
  __builtin_amdgcn_s_wait_asynccnt(0);
#endif
}

// ---------------------------------------------------------------------------
// WMMA helpers: CDNA5 16x16x32 f16 fragments (wave32), contiguous k order k = 16*(lane/16) + s in A and B
// ---------------------------------------------------------------------------
__device__ __forceinline__ v8f wmma_f(v16h a, v16h b, v8f c) {
  return __builtin_amdgcn_wmma_f32_16x16x32_f16(
      /*neg_a=*/false, a, /*neg_b=*/false, b,
      /*c_mod=*/(short)0, c, /*reuse_a=*/false, /*reuse_b=*/false);
}

// A tile 16(M) x 32(K), row-major, row stride lda (elements).
__device__ __forceinline__ v16h load_a16x32(const _Float16* __restrict__ base, int lda) {
  int lane = threadIdx.x & 31;
  const _Float16* row = base + (size_t)(lane & 15) * lda + ((lane >> 4) << 4);
  v16h a;
#pragma unroll
  for (int v = 0; v < 8; ++v) {
    int kb = v << 1;  // 2v
    h2 p = *(const h2*)(row + kb);
    a[2 * v] = p[0];
    a[2 * v + 1] = p[1];
  }
  return a;
}

// B tile 32(K) x 16(N), column-major (column contiguous in K), col stride ldcol.
__device__ __forceinline__ v16h load_b32x16(const _Float16* __restrict__ base, int ldcol) {
  int lane = threadIdx.x & 31;
  const _Float16* col = base + (size_t)(lane & 15) * ldcol + ((lane >> 4) << 4);
  return *(const v16h*)col;
}

// ---------------------------------------------------------------------------
// 1) GroupNorm statistics: one block per (b, group); region is contiguous.
// ---------------------------------------------------------------------------
__global__ void gn_stats_kernel(const float* __restrict__ x,
                                float* __restrict__ mean, float* __restrict__ rstd) {
  int bg = blockIdx.x;  // b*32 + g
  const float* p = x + (size_t)bg * (GSZ * NT);
  float s = 0.f, ss = 0.f;
  for (int i = threadIdx.x; i < GSZ * NT; i += blockDim.x) {
    float v = p[i];
    s += v; ss += v * v;
  }
#pragma unroll
  for (int m = 16; m >= 1; m >>= 1) {
    s  += __shfl_xor(s,  m, 32);
    ss += __shfl_xor(ss, m, 32);
  }
  __shared__ float sh[16];
  int wave = threadIdx.x >> 5;
  if ((threadIdx.x & 31) == 0) { sh[wave * 2] = s; sh[wave * 2 + 1] = ss; }
  __syncthreads();
  if (threadIdx.x == 0) {
    float S = 0.f, SS = 0.f;
#pragma unroll
    for (int w = 0; w < 8; ++w) { S += sh[2 * w]; SS += sh[2 * w + 1]; }
    float mu  = S / (float)(GSZ * NT);
    float var = SS / (float)(GSZ * NT) - mu * mu;
    mean[bg] = mu;
    rstd[bg] = rsqrtf(var + 1e-5f);
  }
}

// ---------------------------------------------------------------------------
// 2) Apply GN + transpose [b][c][n] -> token-major f16 [b][n][c] via LDS tile.
// ---------------------------------------------------------------------------
__global__ void gn_apply_kernel(const float* __restrict__ x,
                                const float* __restrict__ mean, const float* __restrict__ rstd,
                                const float* __restrict__ nw, const float* __restrict__ nb,
                                _Float16* __restrict__ xn) {
  __shared__ _Float16 t[32][33];
  int b = blockIdx.z, c0 = blockIdx.y * 32, n0 = blockIdx.x * 32;
  int col = threadIdx.x & 31, rb = threadIdx.x >> 5;
#pragma unroll
  for (int it = 0; it < 4; ++it) {
    int row = rb + it * 8;
    int c = c0 + row, n = n0 + col;
    int g = (b << 5) + (c >> 4);
    float v = (x[((size_t)b * NC + c) * NT + n] - mean[g]) * rstd[g] * nw[c] + nb[c];
    t[row][col] = (_Float16)v;
  }
  __syncthreads();
#pragma unroll
  for (int it = 0; it < 4; ++it) {
    int nrow = rb + it * 8;
    xn[((size_t)b * NT + n0 + nrow) * NC + c0 + col] = t[col][nrow];
  }
}

// ---------------------------------------------------------------------------
// 3) Weight conversion f32 -> f16.
// ---------------------------------------------------------------------------
__global__ void wconv_kernel(const float* __restrict__ qw, const float* __restrict__ pw,
                             _Float16* __restrict__ qwh, _Float16* __restrict__ pwh) {
  int i = blockIdx.x * blockDim.x + threadIdx.x;
  if (i < 3 * NC * NC) qwh[i] = (_Float16)qw[i];
  if (i < NC * NC)     pwh[i] = (_Float16)pw[i];
}

// ---------------------------------------------------------------------------
// Shared-B GEMM skeleton: block = 8 waves, tile 128(M) x 64(N), K = 512.
// B panel (64 tokens x 32 K = 4KB) staged once per K-step into double-buffered
// LDS (async), consumed by all 8 waves; each wave does 16x64 (4 acc, A reused).
// ---------------------------------------------------------------------------
__device__ __forceinline__ void stage_b_panel(const _Float16* __restrict__ xb,
                                              int n0, int c0, _Float16* buf, int tid) {
  // 64 columns x 64B (32 halves) each = 256 b128 chunks, 256 threads
  int col = tid >> 2, sub = tid & 3;
  async_copy_b128(xb + (size_t)(n0 + col) * NC + c0 + sub * 8, buf + col * 32 + sub * 8);
}

// ---------------------------------------------------------------------------
// 4) QKV GEMM + scatter into attention layouts.
// ---------------------------------------------------------------------------
__global__ void qkv_gemm_kernel(const _Float16* __restrict__ wh, const _Float16* __restrict__ xn,
                                const float* __restrict__ bias,
                                _Float16* __restrict__ qb, _Float16* __restrict__ kb,
                                _Float16* __restrict__ vb) {
  __shared__ __align__(32) _Float16 bsh[2][64 * 32];
  int b = blockIdx.z, tid = threadIdx.x;
  int wave = tid >> 5, lane = tid & 31;
  int o0 = blockIdx.y * 128 + wave * 16;
  int n0 = blockIdx.x * 64;
  const _Float16* xb = xn + (size_t)b * NT * NC;
  v8f acc[4] = {{}, {}, {}, {}};

  stage_b_panel(xb, n0, 0, bsh[0], tid);
  for (int s = 0; s < NC / 32; ++s) {
    int cur = s & 1;
    wait_async();
    __syncthreads();
    if (s + 1 < NC / 32) stage_b_panel(xb, n0, (s + 1) * 32, bsh[cur ^ 1], tid);
    v16h a = load_a16x32(wh + (size_t)o0 * NC + s * 32, NC);
#pragma unroll
    for (int j = 0; j < 4; ++j) {
      v16h bt = load_b32x16(bsh[cur] + j * 16 * 32, 32);
      acc[j] = wmma_f(a, bt, acc[j]);
    }
  }

  int N = lane & 15, Mb = (lane >> 4) << 3;
#pragma unroll
  for (int j = 0; j < 4; ++j) {
    int n = n0 + j * 16 + N;
#pragma unroll
    for (int r = 0; r < 8; ++r) {
      int o = o0 + Mb + r;
      float val = acc[j][r] + bias[o];
      int which = o >> 9;  // 0:q 1:k 2:v (uniform per wave)
      int c = o & 511;
      int h = c >> 7, d = c & 127;
      _Float16 hv = (_Float16)val;
      if (which == 0)      qb[(((size_t)b * NH + h) * NT + n) * HD + d] = hv;
      else if (which == 1) kb[(((size_t)b * NH + h) * NT + n) * HD + d] = hv;
      else                 vb[(((size_t)b * NH + h) * HD + d) * NT + n] = hv;
    }
  }
}

// ---------------------------------------------------------------------------
// 5) Flash attention: block = 4 waves x 16 query rows; K/V tiles (8KB each)
//    shared by all waves, async double-buffered through LDS.
// ---------------------------------------------------------------------------
__global__ void flash_kernel(const _Float16* __restrict__ q, const _Float16* __restrict__ k,
                             const _Float16* __restrict__ v, _Float16* __restrict__ out) {
  __shared__ __align__(32) _Float16 ksh[2][32 * HD];  // [mloc][d]
  __shared__ __align__(32) _Float16 vsh[2][HD * 32];  // [d][mloc]
  __shared__ __align__(32) _Float16 psh[4 * 16 * 32]; // per-wave P staging
  int bh   = blockIdx.y;  // b*NH + h
  int tid  = threadIdx.x;
  int wave = tid >> 5, lane = tid & 31;
  int n0 = blockIdx.x * 64 + wave * 16;
  const _Float16* qh = q + (size_t)bh * NT * HD;
  const _Float16* kh = k + (size_t)bh * NT * HD;
  const _Float16* vh = v + (size_t)bh * HD * NT;

  v16h qa[4];
#pragma unroll
  for (int kt = 0; kt < 4; ++kt)
    qa[kt] = load_a16x32(qh + (size_t)n0 * HD + kt * 32, HD);

  v8f acc[8];
#pragma unroll
  for (int dt = 0; dt < 8; ++dt) acc[dt] = {};
  float mi[8], li[8];
#pragma unroll
  for (int r = 0; r < 8; ++r) { mi[r] = -3.0e38f; li[r] = 0.f; }

  const float scale = 0.08838834764831845f;  // 1/sqrt(128)
  _Float16* pw = psh + wave * (16 * 32);
  int N = lane & 15, Mb = (lane >> 4) << 3;

  // cooperative K/V tile staging: 128 threads, 512+512 b128 chunks
  auto stage_kv = [&](int m0, int bufsel) {
    const _Float16* ksrc = kh + (size_t)m0 * HD;  // 8KB contiguous
    _Float16* kd = ksh[bufsel];
    _Float16* vd = vsh[bufsel];
#pragma unroll
    for (int i = 0; i < 4; ++i) {
      int c = tid + i * 128;
      async_copy_b128(ksrc + c * 8, kd + c * 8);
    }
#pragma unroll
    for (int i = 0; i < 4; ++i) {
      int c = tid + i * 128;            // 0..511
      int row = c >> 2, sub = c & 3;    // d row, 4 chunks of 8 per row
      async_copy_b128(vh + (size_t)row * NT + m0 + sub * 8, vd + row * 32 + sub * 8);
    }
  };

  stage_kv(0, 0);
  for (int s = 0; s < NT / 32; ++s) {
    int cur = s & 1;
    wait_async();
    __syncthreads();
    if (s + 1 < NT / 32) stage_kv((s + 1) * 32, cur ^ 1);
    const _Float16* kc = ksh[cur];
    const _Float16* vc = vsh[cur];

    // S = Q x K^T : two 16x16 f32 tiles
    v8f s0 = {}, s1 = {};
#pragma unroll
    for (int kt = 0; kt < 4; ++kt) {
      v16h b0 = load_b32x16(kc + kt * 32, HD);
      s0 = wmma_f(qa[kt], b0, s0);
      v16h b1 = load_b32x16(kc + 16 * HD + kt * 32, HD);
      s1 = wmma_f(qa[kt], b1, s1);
    }
    // online softmax (row reductions across 16-lane half-wave)
    float p0[8], p1[8], alpha[8];
#pragma unroll
    for (int r = 0; r < 8; ++r) {
      float a0 = s0[r] * scale, a1 = s1[r] * scale;
      float mx = fmaxf(a0, a1);
#pragma unroll
      for (int msk = 8; msk >= 1; msk >>= 1) mx = fmaxf(mx, __shfl_xor(mx, msk, 32));
      float mnew = fmaxf(mi[r], mx);
      alpha[r] = __expf(mi[r] - mnew);
      float e0 = __expf(a0 - mnew), e1 = __expf(a1 - mnew);
      float rs = e0 + e1;
#pragma unroll
      for (int msk = 8; msk >= 1; msk >>= 1) rs += __shfl_xor(rs, msk, 32);
      li[r] = li[r] * alpha[r] + rs;
      mi[r] = mnew;
      p0[r] = e0; p1[r] = e1;
    }
#pragma unroll
    for (int dt = 0; dt < 8; ++dt)
#pragma unroll
      for (int r = 0; r < 8; ++r) acc[dt][r] *= alpha[r];

    // stage P (f16, 16x32 row-major) in wave-private LDS; same-wave DS in-order
#pragma unroll
    for (int r = 0; r < 8; ++r) {
      pw[(Mb + r) * 32 + N]      = (_Float16)p0[r];
      pw[(Mb + r) * 32 + N + 16] = (_Float16)p1[r];
    }
    asm volatile("s_wait_dscnt 0" ::: "memory");
    v16h pa = load_a16x32(pw, 32);

    // out += P x V over 8 d-tiles
#pragma unroll
    for (int dt = 0; dt < 8; ++dt) {
      v16h bv = load_b32x16(vc + dt * 16 * 32, 32);
      acc[dt] = wmma_f(pa, bv, acc[dt]);
    }
  }

#pragma unroll
  for (int r = 0; r < 8; ++r) li[r] = 1.f / li[r];
  int b = bh >> 2, h = bh & 3;
#pragma unroll
  for (int dt = 0; dt < 8; ++dt)
#pragma unroll
    for (int r = 0; r < 8; ++r) {
      int n = n0 + Mb + r;
      int c = h * HD + dt * 16 + N;
      out[((size_t)b * NT + n) * NC + c] = (_Float16)(acc[dt][r] * li[r]);
    }
}

// ---------------------------------------------------------------------------
// 6) Proj GEMM + bias + residual, f32 output (same shared-B skeleton).
// ---------------------------------------------------------------------------
__global__ void proj_gemm_kernel(const _Float16* __restrict__ wh, const _Float16* __restrict__ ah,
                                 const float* __restrict__ bias, const float* __restrict__ x,
                                 float* __restrict__ y) {
  __shared__ __align__(32) _Float16 bsh[2][64 * 32];
  int b = blockIdx.z, tid = threadIdx.x;
  int wave = tid >> 5, lane = tid & 31;
  int o0 = blockIdx.y * 128 + wave * 16;
  int n0 = blockIdx.x * 64;
  const _Float16* ab = ah + (size_t)b * NT * NC;
  v8f acc[4] = {{}, {}, {}, {}};

  stage_b_panel(ab, n0, 0, bsh[0], tid);
  for (int s = 0; s < NC / 32; ++s) {
    int cur = s & 1;
    wait_async();
    __syncthreads();
    if (s + 1 < NC / 32) stage_b_panel(ab, n0, (s + 1) * 32, bsh[cur ^ 1], tid);
    v16h a = load_a16x32(wh + (size_t)o0 * NC + s * 32, NC);
#pragma unroll
    for (int j = 0; j < 4; ++j) {
      v16h bt = load_b32x16(bsh[cur] + j * 16 * 32, 32);
      acc[j] = wmma_f(a, bt, acc[j]);
    }
  }

  int N = lane & 15, Mb = (lane >> 4) << 3;
#pragma unroll
  for (int j = 0; j < 4; ++j) {
    int n = n0 + j * 16 + N;
#pragma unroll
    for (int r = 0; r < 8; ++r) {
      int o = o0 + Mb + r;
      size_t idx = ((size_t)b * NC + o) * NT + n;
      y[idx] = acc[j][r] + bias[o] + x[idx];
    }
  }
}

// ---------------------------------------------------------------------------
extern "C" void kernel_launch(void* const* d_in, const int* in_sizes, int n_in,
                              void* d_out, int out_size, void* d_ws, size_t ws_size,
                              hipStream_t stream) {
  (void)in_sizes; (void)n_in; (void)out_size; (void)ws_size;
  const float* x     = (const float*)d_in[0];
  const float* nw    = (const float*)d_in[1];
  const float* nbias = (const float*)d_in[2];
  const float* qw    = (const float*)d_in[3];
  const float* qbias = (const float*)d_in[4];
  const float* pw    = (const float*)d_in[5];
  const float* pbias = (const float*)d_in[6];
  float* out = (float*)d_out;

  char* ws = (char*)d_ws;
  size_t off = 0;
  auto alloc = [&](size_t bytes) -> void* {
    void* p = ws + off;
    off += (bytes + 255) & ~(size_t)255;
    return p;
  };
  float*     gmean = (float*)alloc(512 * sizeof(float));
  float*     grstd = (float*)alloc(512 * sizeof(float));
  _Float16*  qwh   = (_Float16*)alloc((size_t)3 * NC * NC * 2);
  _Float16*  pwh   = (_Float16*)alloc((size_t)NC * NC * 2);
  _Float16*  xn    = (_Float16*)alloc((size_t)NB * NT * NC * 2);  // reused as attn-out
  _Float16*  qb    = (_Float16*)alloc((size_t)NB * NH * NT * HD * 2);
  _Float16*  kb    = (_Float16*)alloc((size_t)NB * NH * NT * HD * 2);
  _Float16*  vb    = (_Float16*)alloc((size_t)NB * NH * HD * NT * 2);
  _Float16*  ao    = xn;  // xn fully consumed by qkv_gemm before flash writes ao

  gn_stats_kernel<<<NB * 32, 256, 0, stream>>>(x, gmean, grstd);
  gn_apply_kernel<<<dim3(NT / 32, NC / 32, NB), 256, 0, stream>>>(x, gmean, grstd, nw, nbias, xn);
  wconv_kernel<<<(3 * NC * NC) / 256, 256, 0, stream>>>(qw, pw, qwh, pwh);
  qkv_gemm_kernel<<<dim3(NT / 64, (3 * NC) / 128, NB), 256, 0, stream>>>(qwh, xn, qbias, qb, kb, vb);
  flash_kernel<<<dim3(NT / 64, NB * NH), 128, 0, stream>>>(qb, kb, vb, ao);
  proj_gemm_kernel<<<dim3(NT / 64, NC / 128, NB), 256, 0, stream>>>(pwh, ao, pbias, x, out);
}
